// RealSHT_40647570489991
// MI455X (gfx1250) — matrix operand order repaired and checked by reading, not compiled, not run on hardware
//
#include <hip/hip_runtime.h>
#include <math.h>

// Real SHT on gfx1250: both stages as bf16 WMMA GEMMs (f32 accumulate).
// Stage 1: DFT-as-GEMM  Y[j][bc][k] = sum_n x[bc,k,n] * T[n,j]
// Stage 2: per-m GEMM   out[bc,l,m](re/im) = sum_k W[m,l,k] * Y[2m+part][bc][k]

#define NLAT   361
#define NLON   720
#define LMAX   360
#define MMAX   361
#define BATCH  128                 // 4*32
#define R_ROWS (BATCH * NLAT)      // 46208 = 16 * 2888
#define KP1    736                 // nlon padded to mult of 32
#define JP     768                 // 2*MMAX=722 padded to mult of 64
#define KP2    384                 // nlat padded to mult of 32
#define LP     368                 // lmax padded to mult of 16

typedef __attribute__((ext_vector_type(16))) __bf16 v16bf;
typedef __attribute__((ext_vector_type(8)))  float  v8f;

// ---------------- workspace layout (bytes) ----------------
#define WS_XBF  ((size_t)0)                                       // R_ROWS*KP1 bf16 = 68,018,176
#define WS_TT   (WS_XBF + (size_t)R_ROWS * KP1 * 2)               // JP*KP1 bf16     =  1,130,496
#define WS_Y    (WS_TT  + (size_t)JP * KP1 * 2)                   // JP*BATCH*KP2    = 75,497,472
#define WS_W    (WS_Y   + (size_t)JP * BATCH * KP2 * 2)           // MMAX*LP*KP2     =102,027,264

// ---------------- prep kernels ----------------
__global__ __launch_bounds__(256) void k_prep_x(const float* __restrict__ x,
                                                __bf16* __restrict__ xbf) {
    size_t idx = (size_t)blockIdx.x * 256 + threadIdx.x;
    size_t N = (size_t)R_ROWS * KP1;
    if (idx >= N) return;
    int n = (int)(idx % KP1);
    size_t r = idx / KP1;
    float v = (n < NLON) ? x[r * NLON + n] : 0.0f;
    xbf[idx] = (__bf16)v;
}

__global__ __launch_bounds__(256) void k_prep_tw(__bf16* __restrict__ tt) {
    size_t idx = (size_t)blockIdx.x * 256 + threadIdx.x;
    size_t N = (size_t)JP * KP1;
    if (idx >= N) return;
    int n = (int)(idx % KP1);   // longitude sample
    int j = (int)(idx / KP1);   // 2m (+1 for imag)
    float v = 0.0f;
    if (j < 2 * MMAX && n < NLON) {
        int m  = j >> 1;
        int mm = (int)(((long long)m * n) % NLON);     // exact angle reduction
        float theta = (float)mm * (6.28318530717958647692f / NLON);
        float s, c;
        __sincosf(theta, &s, &c);
        float scale = 6.28318530717958647692f / NLON;  // 2*pi * (1/N forward norm)
        v = (j & 1) ? (-scale * s) : (scale * c);
    }
    tt[idx] = (__bf16)v;   // stored transposed: tt[j*KP1 + n]
}

__global__ __launch_bounds__(256) void k_prep_w(const float* __restrict__ w,
                                                __bf16* __restrict__ wbf) {
    size_t idx = (size_t)blockIdx.x * 256 + threadIdx.x;
    size_t N = (size_t)MMAX * LP * KP2;
    if (idx >= N) return;
    int k = (int)(idx % KP2);
    size_t t = idx / KP2;
    int l = (int)(t % LP);
    int m = (int)(t / LP);
    float v = (l < LMAX && k < NLAT) ? w[((size_t)m * LMAX + l) * NLAT + k] : 0.0f;
    wbf[idx] = (__bf16)v;
}

__global__ __launch_bounds__(256) void k_zero_ypad(__bf16* __restrict__ y) {
    size_t idx = (size_t)blockIdx.x * 256 + threadIdx.x;
    size_t N = (size_t)JP * BATCH * (KP2 - NLAT);
    if (idx >= N) return;
    int kk = (int)(idx % (KP2 - NLAT));
    size_t t = idx / (KP2 - NLAT);
    int bc = (int)(t % BATCH);
    int j  = (int)(t / BATCH);
    y[((size_t)j * BATCH + bc) * KP2 + NLAT + kk] = (__bf16)0.0f;
}

// ---------------- stage 1: DFT GEMM ----------------
// grid (R_ROWS/32, JP/64), 256 thr = 8 waves in 2x4; wave tile 16x16, K=736
__global__ __launch_bounds__(256) void k_gemm1(const __bf16* __restrict__ xbf,
                                               const __bf16* __restrict__ tt,
                                               __bf16* __restrict__ y) {
    const int lane = threadIdx.x & 31;
    const int wave = threadIdx.x >> 5;
    const int tm = blockIdx.x * 2 + (wave >> 2);
    const int tn = blockIdx.y * 4 + (wave & 3);
    const int mrow = tm * 16 + (lane & 15);        // r row
    const int jcol = tn * 16 + (lane & 15);        // j col
    const int abase = (lane < 16) ? 0 : 8;
    const int kbase = (lane < 16) ? 0 : 8;
    const __bf16* pa = xbf + (size_t)mrow * KP1 + abase;
    const __bf16* pb = tt  + (size_t)jcol * KP1 + kbase;

    v8f acc = {};
    for (int kk = 0; kk < KP1; kk += 32) {
        v16bf a, b;
        ((uint4*)&a)[0] = *(const uint4*)(pa + kk);        // A: K=base..base+7
        ((uint4*)&a)[1] = *(const uint4*)(pa + kk + 16);   // A: K=16+base..
        ((uint4*)&b)[0] = *(const uint4*)(pb + kk);        // B: K=kbase..kbase+7
        ((uint4*)&b)[1] = *(const uint4*)(pb + kk + 16);   // B: K=16+kbase..
        acc = __builtin_amdgcn_wmma_f32_16x16x32_bf16(false, a, false, b,
                                                      (short)0, acc, false, false);
    }
    const int hi8 = (lane < 16) ? 0 : 8;
    #pragma unroll
    for (int i = 0; i < 8; ++i) {
        int r  = tm * 16 + i + hi8;
        int bc = r / NLAT;
        int k  = r - bc * NLAT;
        y[((size_t)jcol * BATCH + bc) * KP2 + k] = (__bf16)acc[i];
    }
}

// ---------------- stage 2: per-m Legendre GEMM ----------------
// grid (LP/16, 2*MMAX), 256 thr = 8 waves, wave = bc tile; K=384
__global__ __launch_bounds__(256) void k_gemm2(const __bf16* __restrict__ wbf,
                                               const __bf16* __restrict__ y,
                                               float* __restrict__ out) {
    const int lane = threadIdx.x & 31;
    const int wave = threadIdx.x >> 5;
    const int tl = blockIdx.x;           // l tile
    const int j  = blockIdx.y;           // 0..721
    const int m = j >> 1, part = j & 1;
    const int lrow  = tl * 16 + (lane & 15);
    const int bccol = wave * 16 + (lane & 15);
    const int abase = (lane < 16) ? 0 : 8;
    const int kbase = (lane < 16) ? 0 : 8;
    const __bf16* pa = wbf + ((size_t)m * LP + lrow) * KP2 + abase;
    const __bf16* pb = y   + ((size_t)j * BATCH + bccol) * KP2 + kbase;

    v8f acc = {};
    for (int kk = 0; kk < KP2; kk += 32) {
        v16bf a, b;
        ((uint4*)&a)[0] = *(const uint4*)(pa + kk);
        ((uint4*)&a)[1] = *(const uint4*)(pa + kk + 16);
        ((uint4*)&b)[0] = *(const uint4*)(pb + kk);
        ((uint4*)&b)[1] = *(const uint4*)(pb + kk + 16);
        acc = __builtin_amdgcn_wmma_f32_16x16x32_bf16(false, a, false, b,
                                                      (short)0, acc, false, false);
    }
    const int hi8 = (lane < 16) ? 0 : 8;
    #pragma unroll
    for (int i = 0; i < 8; ++i) {
        int l = tl * 16 + i + hi8;
        if (l < LMAX) {
            out[(((size_t)bccol * LMAX + l) * MMAX + m) * 2 + part] = acc[i];
        }
    }
}

extern "C" void kernel_launch(void* const* d_in, const int* in_sizes, int n_in,
                              void* d_out, int out_size, void* d_ws, size_t ws_size,
                              hipStream_t stream) {
    const float* x = (const float*)d_in[0];   // [4,32,361,720] f32
    const float* w = (const float*)d_in[1];   // [361,360,361]  f32
    float* out = (float*)d_out;               // complex64 as interleaved f32 pairs

    char* ws = (char*)d_ws;
    __bf16* xbf = (__bf16*)(ws + WS_XBF);
    __bf16* tt  = (__bf16*)(ws + WS_TT);
    __bf16* y   = (__bf16*)(ws + WS_Y);
    __bf16* wbf = (__bf16*)(ws + WS_W);

    {   // prep: x -> bf16 padded
        size_t n = (size_t)R_ROWS * KP1;
        k_prep_x<<<dim3((unsigned)((n + 255) / 256)), dim3(256), 0, stream>>>(x, xbf);
    }
    {   // prep: twiddle matrix (transposed, bf16)
        size_t n = (size_t)JP * KP1;
        k_prep_tw<<<dim3((unsigned)((n + 255) / 256)), dim3(256), 0, stream>>>(tt);
    }
    {   // prep: weights -> bf16 padded
        size_t n = (size_t)MMAX * LP * KP2;
        k_prep_w<<<dim3((unsigned)((n + 255) / 256)), dim3(256), 0, stream>>>(w, wbf);
    }
    {   // zero K-pad of intermediate Y
        size_t n = (size_t)JP * BATCH * (KP2 - NLAT);
        k_zero_ypad<<<dim3((unsigned)((n + 255) / 256)), dim3(256), 0, stream>>>(y);
    }
    // stage 1: DFT GEMM  (1444 x 12 blocks, 8 waves each)
    k_gemm1<<<dim3(R_ROWS / 32, JP / 64), dim3(256), 0, stream>>>(xbf, tt, y);
    // stage 2: per-m Legendre GEMM  (23 x 722 blocks, 8 waves each)
    k_gemm2<<<dim3(LP / 16, 2 * MMAX), dim3(256), 0, stream>>>(wbf, y, out);
}
